// PerformerBlock_43808666419233
// MI455X (gfx1250) — matrix operand order repaired and checked by reading, not compiled, not run on hardware
//
#include <hip/hip_runtime.h>
#include <hip/hip_bf16.h>
#include <math.h>

// ---------------- types ----------------
typedef _Float16 half16 __attribute__((ext_vector_type(16)));
typedef _Float16 half8  __attribute__((ext_vector_type(8)));
typedef float    float8 __attribute__((ext_vector_type(8)));
typedef _Float16 h16;

// ---------------- problem constants ----------------
#define CB   4
#define CN   4096
#define CDIM 1024
#define CH   16
#define CD   64
#define CM   256
#define CBN  (CB * CN)      // 16384
#define CBH  (CB * CH)      // 64

// ---------------- WMMA core ----------------
__device__ __forceinline__ float8 wmma_f16(half16 a, half16 b, float8 c) {
  return __builtin_amdgcn_wmma_f32_16x16x32_f16(
      /*neg_a=*/false, a, /*neg_b=*/false, b,
      /*c_mod=*/(short)0, c, /*reuse_a=*/false, /*reuse_b=*/false);
}

// ---- fragment loads straight from global (used by small-K GEMMs) ----
__device__ __forceinline__ half16 load_a_frag(const h16* __restrict__ A, int lda,
                                              int row0, int kb, int lane) {
  const int r  = row0 + (lane & 15);
  const int hi = (lane >> 4) & 1;
  const h16* p = A + (size_t)r * (size_t)lda + kb + hi * 8;
  half8 lo = *(const half8*)(p);
  half8 hh = *(const half8*)(p + 16);
  half16 out;
#pragma unroll
  for (int i = 0; i < 8; ++i) { out[i] = lo[i]; out[i + 8] = hh[i]; }
  return out;
}
__device__ __forceinline__ half16 load_b_frag(const h16* __restrict__ Bt, int ldb,
                                              int col0, int kb, int lane) {
  const int c  = col0 + (lane & 15);
  const int hi = (lane >> 4) & 1;
  const h16* p = Bt + (size_t)c * (size_t)ldb + kb + hi * 8;
  half8 lo = *(const half8*)(p);
  half8 hh = *(const half8*)(p + 16);
  half16 out;
#pragma unroll
  for (int i = 0; i < 8; ++i) { out[i] = lo[i]; out[i + 8] = hh[i]; }
  return out;
}

// ---- fragment loads from padded LDS slab (row stride PADW halves = 80B) ----
#define PADW 40
__device__ __forceinline__ half16 lds_a_frag(const h16* base, int row0, int hi, int l) {
  const h16* p = base + (size_t)(row0 + l) * PADW + hi * 8;
  half8 lo = *(const half8*)(p);
  half8 hh = *(const half8*)(p + 16);
  half16 out;
#pragma unroll
  for (int i = 0; i < 8; ++i) { out[i] = lo[i]; out[i + 8] = hh[i]; }
  return out;
}
__device__ __forceinline__ half16 lds_b_frag(const h16* base, int col0, int hi, int l) {
  const h16* p = base + (size_t)(col0 + l) * PADW + hi * 8;
  half8 lo = *(const half8*)(p);
  half8 hh = *(const half8*)(p + 16);
  half16 out;
#pragma unroll
  for (int i = 0; i < 8; ++i) { out[i] = lo[i]; out[i + 8] = hh[i]; }
  return out;
}

// ---------------- direct (register-streamed) WMMA GEMM, small K ----------------
template <int WM, int WN, class Ep>
__global__ __launch_bounds__(256) void gemm_wmma_kernel(
    const h16* __restrict__ A, const h16* __restrict__ Bt,
    int K, int lda, int ldb, long long strideA, long long strideB, Ep ep) {
  const int z = blockIdx.z;
  A  += (long long)z * strideA;
  Bt += (long long)z * strideB;
  const int wid  = threadIdx.x >> 5;
  const int lane = threadIdx.x & 31;
  const int wm = wid / WN, wn = wid % WN;
  const int m0 = (blockIdx.y * WM + wm) * 32;
  const int n0 = (blockIdx.x * WN + wn) * 32;

  float8 c00 = {}, c01 = {}, c10 = {}, c11 = {};
  for (int kb = 0; kb < K; kb += 32) {
    half16 a0 = load_a_frag(A, lda, m0,      kb, lane);
    half16 a1 = load_a_frag(A, lda, m0 + 16, kb, lane);
    half16 b0 = load_b_frag(Bt, ldb, n0,      kb, lane);
    half16 b1 = load_b_frag(Bt, ldb, n0 + 16, kb, lane);
    c00 = wmma_f16(a0, b0, c00);
    c01 = wmma_f16(a0, b1, c01);
    c10 = wmma_f16(a1, b0, c10);
    c11 = wmma_f16(a1, b1, c11);
  }
  const int hi = (lane >> 4) & 1;
  const int cn = lane & 15;
#pragma unroll
  for (int j = 0; j < 8; ++j) {
    const int r = m0 + j + hi * 8;
    ep(z, r,      n0 + cn,      c00[j]);
    ep(z, r,      n0 + 16 + cn, c01[j]);
    ep(z, r + 16, n0 + cn,      c10[j]);
    ep(z, r + 16, n0 + 16 + cn, c11[j]);
  }
}

// ---------------- async double-buffered LDS WMMA GEMM ----------------
// Block tile BM x BN, K slabs of 32. Global->LDS via GLOBAL_LOAD_ASYNC_TO_LDS_B128
// (ASYNCcnt), one slab in flight: s_wait_asynccnt 3 overlaps copy of slab s+1
// with compute of slab s. 3 async ops per wave per slab in both configs.
template <int WM, int WN, class Ep>
__global__ __launch_bounds__(256) void gemm_wmma_async_kernel(
    const h16* __restrict__ A, const h16* __restrict__ Bt,
    int K, int lda, int ldb, long long strideA, long long strideB, Ep ep) {
  constexpr int BM = WM * 32, BN = WN * 32;
  constexpr int SLABH = (BM + BN) * PADW;         // halves per buffer
  static_assert((BM + BN) / 64 == 3, "wait count assumes 3 async ops/wave/slab");
  __shared__ __align__(16) h16 smem[2 * SLABH];

  const int z = blockIdx.z;
  A  += (long long)z * strideA;
  Bt += (long long)z * strideB;
  const int tid  = threadIdx.x;
  const int wid  = tid >> 5;
  const int lane = tid & 31;
  const int wm = wid / WN, wn = wid % WN;
  const int mblk = blockIdx.y * BM, nblk = blockIdx.x * BN;
  const int m0l = wm * 32, n0l = wn * 32;

  const unsigned ldsbase = (unsigned)(size_t)(&smem[0]);

  auto issue_slab = [&](int kb, int buf) {
    const unsigned abase = ldsbase + (unsigned)buf * (SLABH * 2);
    const unsigned bbase = abase + (unsigned)(BM * PADW * 2);
    for (int q = tid; q < BM * 4; q += 256) {           // A: BM rows x 64B
      const int r = q >> 2, s4 = q & 3;
      const unsigned long long ga =
          (unsigned long long)(size_t)(A + (size_t)(mblk + r) * lda + kb + s4 * 8);
      const unsigned dst = abase + (unsigned)(r * (PADW * 2) + s4 * 16);
      asm volatile("global_load_async_to_lds_b128 %0, %1, off"
                   :: "v"(dst), "v"(ga) : "memory");
    }
    for (int q = tid; q < BN * 4; q += 256) {           // Bt: BN rows x 64B
      const int r = q >> 2, s4 = q & 3;
      const unsigned long long ga =
          (unsigned long long)(size_t)(Bt + (size_t)(nblk + r) * ldb + kb + s4 * 8);
      const unsigned dst = bbase + (unsigned)(r * (PADW * 2) + s4 * 16);
      asm volatile("global_load_async_to_lds_b128 %0, %1, off"
                   :: "v"(dst), "v"(ga) : "memory");
    }
  };

  float8 c00 = {}, c01 = {}, c10 = {}, c11 = {};
  const int nslab = K >> 5;
  issue_slab(0, 0);
  for (int s = 0; s < nslab; ++s) {
    if (s + 1 < nslab) {
      issue_slab((s + 1) * 32, (s + 1) & 1);
      asm volatile("s_wait_asynccnt 0x3" ::: "memory");  // slab s landed
    } else {
      asm volatile("s_wait_asynccnt 0x0" ::: "memory");
    }
    __syncthreads();                                     // all waves' slab visible
    const h16* As = smem + (s & 1) * SLABH;
    const h16* Bs = As + BM * PADW;
    const int hi = (lane >> 4) & 1, l15 = lane & 15;
    half16 a0 = lds_a_frag(As, m0l,      hi, l15);
    half16 a1 = lds_a_frag(As, m0l + 16, hi, l15);
    half16 b0 = lds_b_frag(Bs, n0l,      hi, l15);
    half16 b1 = lds_b_frag(Bs, n0l + 16, hi, l15);
    c00 = wmma_f16(a0, b0, c00);
    c01 = wmma_f16(a0, b1, c01);
    c10 = wmma_f16(a1, b0, c10);
    c11 = wmma_f16(a1, b1, c11);
    __syncthreads();                                     // buffer safe to overwrite
  }

  const int m0 = mblk + m0l, n0 = nblk + n0l;
  const int hi = (lane >> 4) & 1;
  const int cn = lane & 15;
#pragma unroll
  for (int j = 0; j < 8; ++j) {
    const int r = m0 + j + hi * 8;
    ep(z, r,      n0 + cn,      c00[j]);
    ep(z, r,      n0 + 16 + cn, c01[j]);
    ep(z, r + 16, n0 + cn,      c10[j]);
    ep(z, r + 16, n0 + 16 + cn, c11[j]);
  }
}

// ---------------- epilogues ----------------
struct EpQKV {   // split qkv -> q[bh,n,e], k[bh,n,e], v^T[bh,e,n]
  h16 *q, *k, *vt;
  __device__ void operator()(int, int row, int col, float acc) const {
    const int b = row >> 12, n = row & 4095;
    const int part = col >> 10, rem = col & 1023;
    const int hh = rem >> 6, e = rem & 63;
    const long long bh = (long long)(b * CH + hh);
    const h16 v = (h16)acc;
    if (part == 0)      q [(bh * CN + n) * CD + e] = v;
    else if (part == 1) k [(bh * CN + n) * CD + e] = v;
    else                vt[(bh * CD + e) * CN + n] = v;
  }
};
struct EpQP {    // qp[bh, n, m]
  h16* qp;
  __device__ void operator()(int z, int row, int col, float acc) const {
    qp[(long long)z * CN * CM + (long long)row * CM + col] = (h16)acc;
  }
};
struct EpKPT {   // kp^T[bh, m, n]
  h16* kpt;
  __device__ void operator()(int z, int row, int col, float acc) const {
    kpt[(long long)z * CM * CN + (long long)col * CN + row] = (h16)acc;
  }
};
struct EpCtxT {  // ctx^T[bh, e, m]  (GEMM computed ctx[m,e])
  h16* ctxt;
  __device__ void operator()(int z, int row, int col, float acc) const {
    ctxt[(long long)z * CD * CM + (long long)col * CM + row] = (h16)acc;
  }
};
struct EpAttn {  // out[bh,n,e] * d_inv, scattered per reference reshape (b,h,n,e)->(b,n',c)
  const float* dinv;
  h16* attn2;
  __device__ void operator()(int z, int row, int col, float acc) const {
    const int b = z >> 4, hh = z & 15;
    const float v = acc * dinv[(long long)z * CN + row];
    const int np = hh * 256 + (row >> 4);
    const int c  = ((row & 15) << 6) | col;
    attn2[((long long)b * CN + np) * CDIM + c] = (h16)v;
  }
};
struct EpWout {  // x2 = x + attn2@w_out + b_out   (fp32 out)
  const float *x, *bout;
  float* x2;
  __device__ void operator()(int, int row, int col, float acc) const {
    const long long idx = (long long)row * CDIM + col;
    x2[idx] = x[idx] + acc + bout[col];
  }
};
struct EpFFN1 {  // h = gelu(ln2@w1 + b1)  exact erf gelu, store f16
  const float* b1;
  h16* hbuf;
  __device__ void operator()(int, int row, int col, float acc) const {
    const float t = acc + b1[col];
    const float g = 0.5f * t * (1.0f + erff(t * 0.70710678118654752f));
    hbuf[(long long)row * (4 * CDIM) + col] = (h16)g;
  }
};
struct EpFFN2 {  // out = x2 + h@w2 + b2   (fp32 out)
  const float *x2, *b2;
  float* out;
  __device__ void operator()(int, int row, int col, float acc) const {
    const long long idx = (long long)row * CDIM + col;
    out[idx] = x2[idx] + acc + b2[col];
  }
};

// ---------------- helper kernels ----------------
__global__ __launch_bounds__(256) void layernorm_f16_kernel(
    const float* __restrict__ in, const float* __restrict__ g,
    const float* __restrict__ beta, h16* __restrict__ out) {
  __shared__ float s_sum[256], s_sq[256];
  const int row = blockIdx.x;
  const float* p = in + (long long)row * CDIM;
  float v[4], sum = 0.f, sq = 0.f;
#pragma unroll
  for (int i = 0; i < 4; ++i) {
    v[i] = p[threadIdx.x + i * 256];
    sum += v[i]; sq += v[i] * v[i];
  }
  s_sum[threadIdx.x] = sum; s_sq[threadIdx.x] = sq;
  __syncthreads();
  for (int s = 128; s > 0; s >>= 1) {
    if (threadIdx.x < s) {
      s_sum[threadIdx.x] += s_sum[threadIdx.x + s];
      s_sq [threadIdx.x] += s_sq [threadIdx.x + s];
    }
    __syncthreads();
  }
  const float mu   = s_sum[0] * (1.0f / CDIM);
  const float var  = s_sq[0] * (1.0f / CDIM) - mu * mu;
  const float rstd = rsqrtf(var + 1e-5f);
  h16* o = out + (long long)row * CDIM;
#pragma unroll
  for (int i = 0; i < 4; ++i) {
    const int c = threadIdx.x + i * 256;
    o[c] = (h16)((v[i] - mu) * rstd * g[c] + beta[c]);
  }
}

__global__ __launch_bounds__(256) void transpose_cvt_kernel(
    const float* __restrict__ in, h16* __restrict__ out, int R, int C) {
  __shared__ float tile[32][33];
  const int bc = blockIdx.x * 32, br = blockIdx.y * 32;
  const int tx = threadIdx.x & 31, ty = threadIdx.x >> 5;
#pragma unroll
  for (int i = 0; i < 32; i += 8)
    tile[ty + i][tx] = in[(long long)(br + ty + i) * C + bc + tx];
  __syncthreads();
#pragma unroll
  for (int i = 0; i < 32; i += 8)
    out[(long long)(bc + ty + i) * R + br + tx] = (h16)tile[tx][ty + i];
}

__global__ __launch_bounds__(256) void cvt_f16_kernel(const float* __restrict__ in,
                                                      h16* __restrict__ out, int n) {
  const int i = blockIdx.x * 256 + threadIdx.x;
  if (i < n) out[i] = (h16)in[i];
}

__global__ __launch_bounds__(256) void rowsum_f16_kernel(const h16* __restrict__ in,
                                                         float* __restrict__ out, int width) {
  __shared__ float s[256];
  const long long row = blockIdx.x;
  const h16* p = in + row * width;
  float sum = 0.f;
  for (int i = threadIdx.x; i < width; i += 256) sum += (float)p[i];
  s[threadIdx.x] = sum; __syncthreads();
  for (int st = 128; st > 0; st >>= 1) {
    if (threadIdx.x < st) s[threadIdx.x] += s[threadIdx.x + st];
    __syncthreads();
  }
  if (threadIdx.x == 0) out[row] = s[0];
}

__global__ __launch_bounds__(256) void dinv_kernel(const h16* __restrict__ qp,
                                                   const float* __restrict__ ksum,
                                                   float* __restrict__ dinv) {
  __shared__ float sk[CM];
  const int z = blockIdx.y;
  const int n = blockIdx.x * 256 + threadIdx.x;
  sk[threadIdx.x] = ksum[z * CM + threadIdx.x];
  __syncthreads();
  const h16* p = qp + ((long long)z * CN + n) * CM;
  float dot = 0.f;
#pragma unroll 8
  for (int m = 0; m < CM; ++m) dot += (float)p[m] * sk[m];
  dinv[(long long)z * CN + n] = 1.0f / dot;
}

// ---------------- launch ----------------
extern "C" void kernel_launch(void* const* d_in, const int* in_sizes, int n_in,
                              void* d_out, int out_size, void* d_ws, size_t ws_size,
                              hipStream_t stream) {
  const float* x     = (const float*)d_in[0];
  const float* proj  = (const float*)d_in[1];
  const float* ln1_g = (const float*)d_in[2];
  const float* ln1_b = (const float*)d_in[3];
  const float* ln2_g = (const float*)d_in[4];
  const float* ln2_b = (const float*)d_in[5];
  const float* w_qkv = (const float*)d_in[6];
  const float* w_out = (const float*)d_in[7];
  const float* b_out = (const float*)d_in[8];
  const float* w1    = (const float*)d_in[9];
  const float* b1    = (const float*)d_in[10];
  const float* w2    = (const float*)d_in[11];
  const float* b2    = (const float*)d_in[12];
  float* outp = (float*)d_out;
  (void)in_sizes; (void)n_in; (void)out_size; (void)ws_size;

  char* ws = (char*)d_ws;
  size_t off = 0;
  auto take = [&](size_t bytes) -> char* {
    char* p = ws + off;
    off = (off + bytes + 255) & ~(size_t)255;
    return p;
  };

  h16*   xln1  = (h16*)  take((size_t)CBN * CDIM * 2);
  h16*   wqkvT = (h16*)  take((size_t)3 * CDIM * CDIM * 2);       // [3072,1024]
  h16*   projh = (h16*)  take((size_t)CM * CD * 2);               // [256,64]
  h16*   qbuf  = (h16*)  take((size_t)CBH * CN * CD * 2);
  h16*   kbuf  = (h16*)  take((size_t)CBH * CN * CD * 2);
  h16*   vt    = (h16*)  take((size_t)CBH * CD * CN * 2);
  h16*   qp    = (h16*)  take((size_t)CBH * CN * CM * 2);
  h16*   kpt   = (h16*)  take((size_t)CBH * CM * CN * 2);
  float* ksum  = (float*)take((size_t)CBH * CM * 4);
  float* dinv  = (float*)take((size_t)CBH * CN * 4);
  h16*   ctxt  = (h16*)  take((size_t)CBH * CD * CM * 2);
  h16*   attn2 = (h16*)  take((size_t)CBN * CDIM * 2);
  h16*   woutT = (h16*)  take((size_t)CDIM * CDIM * 2);
  float* x2    = (float*)take((size_t)CBN * CDIM * 4);
  h16*   ln2b  = (h16*)  take((size_t)CBN * CDIM * 2);
  h16*   w1T   = (h16*)  take((size_t)4 * CDIM * CDIM * 2);       // [4096,1024]
  h16*   hbuf  = (h16*)  take((size_t)CBN * 4 * CDIM * 2);
  h16*   w2T   = (h16*)  take((size_t)CDIM * 4 * CDIM * 2);       // [1024,4096]

  // ---- weight prep: transpose + f16 convert (Bt layout for the GEMMs) ----
  transpose_cvt_kernel<<<dim3(3 * CDIM / 32, CDIM / 32), 256, 0, stream>>>(w_qkv, wqkvT, CDIM, 3 * CDIM);
  transpose_cvt_kernel<<<dim3(CDIM / 32, CDIM / 32),     256, 0, stream>>>(w_out, woutT, CDIM, CDIM);
  transpose_cvt_kernel<<<dim3(4 * CDIM / 32, CDIM / 32), 256, 0, stream>>>(w1, w1T, CDIM, 4 * CDIM);
  transpose_cvt_kernel<<<dim3(CDIM / 32, 4 * CDIM / 32), 256, 0, stream>>>(w2, w2T, 4 * CDIM, CDIM);
  cvt_f16_kernel<<<dim3((CM * CD + 255) / 256), 256, 0, stream>>>(proj, projh, CM * CD);

  // ---- LN1 ----
  layernorm_f16_kernel<<<dim3(CBN), 256, 0, stream>>>(x, ln1_g, ln1_b, xln1);

  // ---- QKV GEMM: [16384,1024] x [1024,3072], scatter q/k/v^T ----
  gemm_wmma_async_kernel<2, 4, EpQKV><<<dim3(3 * CDIM / 128, CBN / 64, 1), 256, 0, stream>>>(
      xln1, wqkvT, CDIM, CDIM, CDIM, 0LL, 0LL, EpQKV{qbuf, kbuf, vt});

  // ---- FAVOR+ projections: per (b,h) [4096,64] x [64,256]  (K=64: direct path) ----
  gemm_wmma_kernel<2, 4, EpQP><<<dim3(CM / 128, CN / 64, CBH), 256, 0, stream>>>(
      qbuf, projh, CD, CD, CD, (long long)CN * CD, 0LL, EpQP{qp});
  gemm_wmma_kernel<2, 4, EpKPT><<<dim3(CM / 128, CN / 64, CBH), 256, 0, stream>>>(
      kbuf, projh, CD, CD, CD, (long long)CN * CD, 0LL, EpKPT{kpt});

  // ---- ksum[m] = sum_n kp[n,m]  (rows of kp^T) ----
  rowsum_f16_kernel<<<dim3(CBH * CM), 256, 0, stream>>>(kpt, ksum, CN);

  // ---- d_inv[n] = 1 / (qp[n,:] . ksum) ----
  dinv_kernel<<<dim3(CN / 256, CBH), 256, 0, stream>>>(qp, ksum, dinv);

  // ---- ctx = kp^T @ v : per (b,h) [256,4096] x [4096,64] -> ctx^T[e,m] ----
  gemm_wmma_async_kernel<4, 2, EpCtxT><<<dim3(CD / 64, CM / 128, CBH), 256, 0, stream>>>(
      kpt, vt, CN, CN, CN, (long long)CM * CN, (long long)CD * CN, EpCtxT{ctxt});

  // ---- out = (qp @ ctx) * d_inv : per (b,h) [4096,256] x [256,64], scatter-reshape ----
  gemm_wmma_async_kernel<4, 2, EpAttn><<<dim3(CD / 64, CN / 128, CBH), 256, 0, stream>>>(
      qp, ctxt, CM, CM, CM, (long long)CN * CM, (long long)CD * CM, EpAttn{dinv, attn2});

  // ---- x2 = x + attn2 @ w_out + b_out ----
  gemm_wmma_async_kernel<2, 4, EpWout><<<dim3(CDIM / 128, CBN / 64, 1), 256, 0, stream>>>(
      attn2, woutT, CDIM, CDIM, CDIM, 0LL, 0LL, EpWout{x, b_out, x2});

  // ---- LN2 ----
  layernorm_f16_kernel<<<dim3(CBN), 256, 0, stream>>>(x2, ln2_g, ln2_b, ln2b);

  // ---- FFN1: gelu(ln2 @ w1 + b1) ----
  gemm_wmma_async_kernel<2, 4, EpFFN1><<<dim3(4 * CDIM / 128, CBN / 64, 1), 256, 0, stream>>>(
      ln2b, w1T, CDIM, CDIM, CDIM, 0LL, 0LL, EpFFN1{b1, hbuf});

  // ---- FFN2: out = x2 + h @ w2 + b2 ----
  gemm_wmma_async_kernel<2, 4, EpFFN2><<<dim3(CDIM / 128, CBN / 64, 1), 256, 0, stream>>>(
      hbuf, w2T, 4 * CDIM, 4 * CDIM, 4 * CDIM, 0LL, 0LL, EpFFN2{x2, b2, outp});
}
